// WD_LSTM_91225105367425
// MI455X (gfx1250) — matrix operand order repaired and checked by reading, not compiled, not run on hardware
//
#include <hip/hip_runtime.h>
#include <cstdint>
#include <cstddef>

// ---------------- problem constants (AWD-LSTM reference) ----------------
#define TT    70
#define BB    80
#define NINP  400
#define NHID  1150
#define NTOK  33278
#define NINP_P 448          // 400 padded to multiple of 64 (BK)

typedef __bf16 bf16_t;
typedef __bf16 v8bf  __attribute__((ext_vector_type(8)));
typedef __bf16 v16bf __attribute__((ext_vector_type(16)));
typedef float  v8f   __attribute__((ext_vector_type(8)));
typedef int    v4i   __attribute__((ext_vector_type(4)));
typedef unsigned uv4 __attribute__((ext_vector_type(4)));
typedef int    iv8   __attribute__((ext_vector_type(8)));
typedef int    iv4   __attribute__((ext_vector_type(4)));

// CDNA5 async global->LDS path (gfx1250) — confirmed present in this toolchain.
#if __has_builtin(__builtin_amdgcn_global_load_async_to_lds_b128)
#define HAVE_ASYNC_LDS 1
#else
#define HAVE_ASYNC_LDS 0
#endif

// CDNA5 Tensor Data Mover (6-arg form in this toolchain). Guarded.
#if __has_builtin(__builtin_amdgcn_tensor_load_to_lds) && \
    __has_builtin(__builtin_amdgcn_s_wait_tensorcnt)
#define HAVE_TDM 1
#else
#define HAVE_TDM 0
#endif

__device__ __forceinline__ void async_wait0() {
#if HAVE_ASYNC_LDS
#if __has_builtin(__builtin_amdgcn_s_wait_asynccnt)
    __builtin_amdgcn_s_wait_asynccnt(0);
#else
    asm volatile("s_wait_asynccnt 0" ::: "memory");
#endif
#endif
}

__device__ __forceinline__ void tensor_wait0() {
#if HAVE_TDM
    __builtin_amdgcn_s_wait_tensorcnt(0);   // no-op for waves with TENSORcnt==0
#endif
}

// Copy 16 bytes (8 bf16) global -> LDS asynchronously (addresses pre-clamped).
__device__ __forceinline__ void stage16(bf16_t* dst, const bf16_t* src) {
#if HAVE_ASYNC_LDS
    __builtin_amdgcn_global_load_async_to_lds_b128(
        (__attribute__((address_space(1))) v4i*)src,
        (__attribute__((address_space(3))) v4i*)dst, 0, 0);
#else
    *(uint4*)dst = *(const uint4*)src;
#endif
}

#if HAVE_TDM
// Issue one TDM DMA of a 2D tile (tile_d0 x tile_d1 elements of 2 bytes) from a
// row-major tensor (row stride = stride0 elements) into contiguous LDS.
// TDM zero-fills reads beyond tensor_d0/tensor_d1 (handles edge tiles).
__device__ __forceinline__ void tdm_load_2d(unsigned lds_addr, unsigned long long gaddr,
                                            unsigned tensor_d0, unsigned tensor_d1,
                                            unsigned long long stride0,
                                            unsigned tile_d0, unsigned tile_d1)
{
    uv4 g0;
    g0[0] = 1u;                                                  // count=1, user D#
    g0[1] = lds_addr;                                            // LDS byte address
    g0[2] = (unsigned)(gaddr & 0xFFFFFFFFull);                   // global addr lo
    g0[3] = (unsigned)((gaddr >> 32) & 0x1FFFFFFull) | (2u << 30); // addr hi | type=2
    iv8 g1;
    g1[0] = (int)(1u << 16);                                     // data_size = 2B
    g1[1] = (int)((tensor_d0 & 0xFFFFu) << 16);                  // tensor_dim0 lo
    g1[2] = (int)((tensor_d0 >> 16) | ((tensor_d1 & 0xFFFFu) << 16));
    g1[3] = (int)((tensor_d1 >> 16) | (tile_d0 << 16));          // tile_dim0
    g1[4] = (int)tile_d1;                                        // tile_dim1, dim2=0
    g1[5] = (int)(unsigned)(stride0 & 0xFFFFFFFFull);            // dim0 stride lo
    g1[6] = (int)(unsigned)((stride0 >> 32) & 0xFFFFu);          // stride hi, s1=0
    g1[7] = 0;
    iv4 z4 = {0, 0, 0, 0};
    iv8 z8 = {0, 0, 0, 0, 0, 0, 0, 0};
    __builtin_amdgcn_tensor_load_to_lds(g0, g1, z4, z4, z8, 0);
}
#endif

// ---------------- deterministic hash RNG for dropout masks ----------------
__device__ __forceinline__ unsigned mixh(unsigned x) {
    x ^= x >> 16; x *= 0x7feb352du;
    x ^= x >> 15; x *= 0x846ca68bu;
    x ^= x >> 16; return x;
}
__device__ __forceinline__ float keep_scale(unsigned seed, unsigned idx, float pkeep) {
    unsigned r = mixh(seed * 0x9E3779B9u + idx * 0x85EBCA6Bu + 0x27220A95u);
    float u = (float)(r >> 8) * (1.0f / 16777216.0f);
    return (u < pkeep) ? (1.0f / pkeep) : 0.0f;
}

__device__ __forceinline__ float fast_tanhf(float x) {
#if __has_builtin(__builtin_amdgcn_tanhf)
    return __builtin_amdgcn_tanhf(x);      // gfx1250 V_TANH_F32
#else
    return 1.0f - 2.0f / (__expf(2.0f * x) + 1.0f);
#endif
}
__device__ __forceinline__ float sigmoidf(float x) {
    return 0.5f * fast_tanhf(0.5f * x) + 0.5f;
}

// =====================================================================
// Tiled bf16 WMMA GEMM:  C[M,N] = A[M,K] * Bt[N,K]^T  (+ bias[N])
// K must be a multiple of 64. Double-buffered LDS. A tile: per-lane
// async global->LDS (clamped rows, no divergence). B tile: one TDM DMA
// issued by wave 0 (zero-filled at the N edge by the tensor engine).
// Tile k+1 streams while tile k runs through v_wmma.
// =====================================================================
template<int WROWS, int WCOLS, int WM, int WN>
__global__ __launch_bounds__(WROWS * WCOLS * 32)
void awd_gemm_bf16_wmma(const bf16_t* __restrict__ A, int lda,
                        const bf16_t* __restrict__ Bt, int ldb,
                        const float* __restrict__ bias,
                        float* __restrict__ C, int ldc,
                        int M, int N, int K)
{
    constexpr int BK = 64;                  // two 32-deep WMMA steps per tile
    constexpr int BM = WROWS * WM * 16;
    constexpr int BN = WCOLS * WN * 16;
    constexpr int NT = WROWS * WCOLS * 32;
    constexpr int CH_A = BM * (BK / 8);     // 16B chunks per A tile
    constexpr int CH_B = BN * (BK / 8);
    constexpr int ITA = (CH_A + NT - 1) / NT;
    constexpr int ITB = (CH_B + NT - 1) / NT;
    constexpr bool AX = (CH_A % NT) == 0;   // staging loop exact -> no guard
    constexpr bool BX = (CH_B % NT) == 0;

    __shared__ __align__(16) bf16_t As[2][BM][BK];
    __shared__ __align__(16) bf16_t Bs[2][BN][BK];

    const int tid   = threadIdx.x;
    const int lane  = tid & 31;
    const int wave  = tid >> 5;
    const int wrow  = wave % WROWS;
    const int wcol  = wave / WROWS;
    const int l15   = lane & 15;
    const int lhalf = lane >> 4;

    const int m0 = blockIdx.y * BM;
    const int n0 = blockIdx.x * BN;

    // ---- hoisted staging descriptors (row-clamped; k-loop just adds k0) ----
    const bf16_t* aSrc[ITA]; bf16_t* aDst[ITA]; bool aVal[ITA];
#pragma unroll
    for (int it = 0; it < ITA; ++it) {
        int c = tid + it * NT;
        bool valid = AX || (c < CH_A);
        int r  = valid ? (c >> 3) : 0;
        int kc = (c & 7) * 8;
        int gm = m0 + r; if (gm >= M) gm = M - 1;   // clamp: row masked at store
        aVal[it] = valid;
        aSrc[it] = A + (size_t)gm * lda + kc;
        aDst[it] = &As[0][r][kc];
    }
#if HAVE_TDM
    const unsigned ldsB =
        (unsigned)(size_t)(__attribute__((address_space(3))) bf16_t*)&Bs[0][0][0];
    const unsigned long long gB =
        (unsigned long long)(size_t)Bt + (size_t)n0 * ldb * 2ull;
#else
    const bf16_t* bSrc[ITB]; bf16_t* bDst[ITB]; bool bVal[ITB];
#pragma unroll
    for (int it = 0; it < ITB; ++it) {
        int c = tid + it * NT;
        bool valid = BX || (c < CH_B);
        int r  = valid ? (c >> 3) : 0;
        int kc = (c & 7) * 8;
        int gn = n0 + r; if (gn >= N) gn = N - 1;   // clamp: col masked at store
        bVal[it] = valid;
        bSrc[it] = Bt + (size_t)gn * ldb + kc;
        bDst[it] = &Bs[0][r][kc];
    }
#endif

    auto stage_tile = [&](int k0, int buf) {
#pragma unroll
        for (int it = 0; it < ITA; ++it)
            if (AX || aVal[it]) stage16(aDst[it] + buf * (BM * BK), aSrc[it] + k0);
#if HAVE_TDM
        if (tid < 32)   // one wave drives the tensor engine
            tdm_load_2d(ldsB + (unsigned)buf * (BN * BK * 2),
                        gB + (unsigned long long)k0 * 2ull,
                        (unsigned)(ldb - k0), (unsigned)(N - n0),
                        (unsigned long long)ldb, (unsigned)BK, (unsigned)BN);
#else
#pragma unroll
        for (int it = 0; it < ITB; ++it)
            if (BX || bVal[it]) stage16(bDst[it] + buf * (BN * BK), bSrc[it] + k0);
#endif
    };

    v8f acc[WM][WN];
    const v8f vzero = {0.f, 0.f, 0.f, 0.f, 0.f, 0.f, 0.f, 0.f};
#pragma unroll
    for (int i = 0; i < WM; ++i)
#pragma unroll
        for (int j = 0; j < WN; ++j) acc[i][j] = vzero;

    const int nk = K / BK;      // K % 64 == 0 by construction
    stage_tile(0, 0);
    async_wait0();
    tensor_wait0();
    __syncthreads();

    for (int ki = 0; ki < nk; ++ki) {
        const int buf = ki & 1;
        if (ki + 1 < nk) stage_tile((ki + 1) * BK, buf ^ 1);   // overlap with WMMA

        // ---- compute: two 32-deep WMMA steps from the current LDS tile ----
#pragma unroll
        for (int s = 0; s < 2; ++s) {
            const int ko = s * 32;
            v16bf afrag[WM], bfrag[WN];
            // A 16x32 bf16: lanes 0-15 = M 0..15 holding K 0..7 & 16..23;
            // lanes 16-31 hold K 8..15 & 24..31.
#pragma unroll
            for (int i = 0; i < WM; ++i) {
                int row = wrow * WM * 16 + i * 16 + l15;
                v8bf lo = *(const v8bf*)&As[buf][row][ko + lhalf * 8];
                v8bf hi = *(const v8bf*)&As[buf][row][ko + 16 + lhalf * 8];
#pragma unroll
                for (int e = 0; e < 8; ++e) { afrag[i][e] = lo[e]; afrag[i][8 + e] = hi[e]; }
            }
            // B 32x16 bf16: lanes 0-15 = N, K 0..7 & 16..23; lanes 16-31 K 8..15 & 24..31.
#pragma unroll
            for (int j = 0; j < WN; ++j) {
                int col = wcol * WN * 16 + j * 16 + l15;
                v8bf lo = *(const v8bf*)&Bs[buf][col][ko + lhalf * 8];
                v8bf hi = *(const v8bf*)&Bs[buf][col][ko + 16 + lhalf * 8];
#pragma unroll
                for (int e = 0; e < 8; ++e) { bfrag[j][e] = lo[e]; bfrag[j][8 + e] = hi[e]; }
            }
#pragma unroll
            for (int i = 0; i < WM; ++i)
#pragma unroll
                for (int j = 0; j < WN; ++j)
                    acc[i][j] = __builtin_amdgcn_wmma_f32_16x16x32_bf16(
                        false, afrag[i], false, bfrag[j], (short)0, acc[i][j], false, false);
        }

        async_wait0();      // next tile streamed during compute
        tensor_wait0();
        __syncthreads();
    }

    // ---- store C (f32): vgpr v -> M = base + v (+8 for upper half-lanes) ----
#pragma unroll
    for (int j = 0; j < WN; ++j) {
        int col = n0 + wcol * WN * 16 + j * 16 + l15;
        if (col >= N) continue;
        const float bv = bias ? bias[col] : 0.0f;
#pragma unroll
        for (int i = 0; i < WM; ++i) {
            int mb = m0 + wrow * WM * 16 + i * 16 + lhalf * 8;
#pragma unroll
            for (int v = 0; v < 8; ++v) {
                int m = mb + v;
                if (m < M) C[(size_t)m * ldc + col] = acc[i][j][v] + bv;
            }
        }
    }
}

// =====================================================================
// Small helper kernels
// =====================================================================

__global__ __launch_bounds__(256)
void awd_cast_pad_kernel(const float* __restrict__ W, bf16_t* __restrict__ out,
                         int rows, int cols, int colsPad, unsigned dropSeed, float pkeep)
{
    long long total = (long long)rows * colsPad;
    for (long long i = (long long)blockIdx.x * 256 + threadIdx.x; i < total;
         i += (long long)gridDim.x * 256) {
        int r = (int)(i / colsPad), k = (int)(i - (long long)r * colsPad);
        float v = 0.0f;
        if (k < cols) {
            v = W[(size_t)r * cols + k];
            if (pkeep < 1.0f) v *= keep_scale(dropSeed, (unsigned)(r * cols + k), pkeep);
        }
        out[i] = (bf16_t)v;
    }
}

__global__ __launch_bounds__(256)
void awd_bias_kernel(const float* __restrict__ bi, const float* __restrict__ bh,
                     float* __restrict__ out, int n, int npad)
{
    int i = blockIdx.x * 256 + threadIdx.x;
    if (i < npad) out[i] = (i < n) ? (bi[i] + bh[i]) : 0.0f;
}

__global__ __launch_bounds__(256)
void awd_embed_kernel(const int* __restrict__ tokens, const float* __restrict__ embW,
                      bf16_t* __restrict__ x)
{
    const int total = TT * BB * NINP_P;
    int idx = blockIdx.x * 256 + threadIdx.x;
    if (idx >= total) return;
    int row = idx / NINP_P, j = idx - row * NINP_P;
    if (j >= NINP) { x[idx] = (bf16_t)0.0f; return; }
    int tok = tokens[row];
    int b = row % BB;
    float e = embW[(size_t)tok * NINP + j];
    e *= keep_scale(0xE111u, (unsigned)tok, 0.90f);              // P_E = 0.1 (whole rows)
    e *= keep_scale(0x1A2Bu, (unsigned)(b * NINP + j), 0.35f);   // P_I = 0.65 (locked)
    x[idx] = (bf16_t)e;
}

__global__ __launch_bounds__(256)
void awd_init_state_kernel(const float* __restrict__ h0, const float* __restrict__ c0,
                           bf16_t* __restrict__ hbf, float* __restrict__ c,
                           int dh, int dhPad)
{
    int i = blockIdx.x * 256 + threadIdx.x;
    if (i >= BB * dhPad) return;
    int b = i / dhPad, j = i - b * dhPad;
    if (j < dh) { hbf[i] = (bf16_t)h0[b * dh + j]; c[i] = c0[b * dh + j]; }
    else        { hbf[i] = (bf16_t)0.0f;           c[i] = 0.0f; }
}

// gates: g = xg[t] + h@W_hh^T (gbuf); update c,h; emit bf16 h and dropped output
__global__ __launch_bounds__(256)
void awd_lstm_gate_kernel(const float* __restrict__ xg_t, const float* __restrict__ gb,
                          float* __restrict__ c, bf16_t* __restrict__ hbf,
                          bf16_t* __restrict__ xnext_t,
                          int dh, int dhPad, int ldg, float pkeep, unsigned seed)
{
    int i = blockIdx.x * 256 + threadIdx.x;
    if (i >= BB * dhPad) return;
    int b = i / dhPad, j = i - b * dhPad;
    if (j >= dh) { xnext_t[i] = (bf16_t)0.0f; return; }
    size_t rb = (size_t)b * ldg;
    float gi = xg_t[rb + j]          + gb[rb + j];
    float gf = xg_t[rb + dh + j]     + gb[rb + dh + j];
    float gg = xg_t[rb + 2 * dh + j] + gb[rb + 2 * dh + j];
    float go = xg_t[rb + 3 * dh + j] + gb[rb + 3 * dh + j];
    float cv = sigmoidf(gf) * c[i] + sigmoidf(gi) * fast_tanhf(gg);
    float hv = sigmoidf(go) * fast_tanhf(cv);
    c[i] = cv;
    hbf[i] = (bf16_t)hv;
    xnext_t[i] = (bf16_t)(hv * keep_scale(seed, (unsigned)i, pkeep));  // locked mask (b,j)
}

// =====================================================================
// Host orchestration
// =====================================================================
extern "C" void kernel_launch(void* const* d_in, const int* in_sizes, int n_in,
                              void* d_out, int out_size, void* d_ws, size_t ws_size,
                              hipStream_t stream)
{
    (void)in_sizes; (void)n_in; (void)out_size; (void)ws_size;

    const int*   tokens = (const int*)d_in[0];
    const float* embW   = (const float*)d_in[1];
    const float* W_ih[3] = {(const float*)d_in[2],  (const float*)d_in[8],  (const float*)d_in[14]};
    const float* W_hh[3] = {(const float*)d_in[3],  (const float*)d_in[9],  (const float*)d_in[15]};
    const float* b_ih[3] = {(const float*)d_in[4],  (const float*)d_in[10], (const float*)d_in[16]};
    const float* b_hh[3] = {(const float*)d_in[5],  (const float*)d_in[11], (const float*)d_in[17]};
    const float* h0[3]   = {(const float*)d_in[6],  (const float*)d_in[12], (const float*)d_in[18]};
    const float* c0[3]   = {(const float*)d_in[7],  (const float*)d_in[13], (const float*)d_in[19]};
    const float* dec_b   = (const float*)d_in[20];
    float* outp = (float*)d_out;

    const int din[3]  = {NINP, NHID, NHID};
    const int dinP[3] = {NINP_P, 1152, 1152};  // K padded to multiple of 64
    const int dh[3]   = {NHID, NHID, NINP};
    const int dhP[3]  = {1152, 1152, NINP_P};
    const int N4[3]   = {4 * NHID, 4 * NHID, 4 * NINP};  // 4600,4600,1600
    const float pkeepOut[3] = {0.70f, 0.70f, 0.60f};     // P_H,P_H,P_DROP
    const int M = TT * BB;                                // 5600
    const int LDG = 4608;                                 // padded 4*dh stride

    auto cdiv = [](long long a, long long b) { return (int)((a + b - 1) / b); };

    // ---- workspace bump allocator (256B aligned) ----
    char* base = (char*)d_ws; size_t off = 0;
    auto alloc = [&](size_t bytes) -> void* {
        void* p = base + off; off = (off + bytes + 255) & ~(size_t)255; return p;
    };
    bf16_t* wih_bf[3]; bf16_t* whh_bf[3]; float* bsum[3];
    for (int l = 0; l < 3; ++l) wih_bf[l] = (bf16_t*)alloc((size_t)N4[l] * dinP[l] * 2);
    for (int l = 0; l < 3; ++l) whh_bf[l] = (bf16_t*)alloc((size_t)N4[l] * dhP[l] * 2);
    for (int l = 0; l < 3; ++l) bsum[l]   = (float*)alloc((size_t)LDG * 4);
    bf16_t* emb_bf = (bf16_t*)alloc((size_t)NTOK * NINP_P * 2);
    bf16_t* xA  = (bf16_t*)alloc((size_t)M * 1152 * 2);
    bf16_t* xB  = (bf16_t*)alloc((size_t)M * 1152 * 2);
    float*  xg  = (float*)alloc((size_t)M * LDG * 4);
    float*  gbf = (float*)alloc((size_t)BB * LDG * 4);
    float*  cst = (float*)alloc((size_t)BB * 1152 * 4);
    bf16_t* hbf = (bf16_t*)alloc((size_t)BB * 1152 * 2);

    // ---- weight preparation (bf16 + pad + DropConnect on W_hh) ----
    for (int l = 0; l < 3; ++l) {
        long long n = (long long)N4[l] * dinP[l];
        awd_cast_pad_kernel<<<cdiv(n, 256), 256, 0, stream>>>(
            W_ih[l], wih_bf[l], N4[l], din[l], dinP[l], 0u, 1.0f);
        n = (long long)N4[l] * dhP[l];
        awd_cast_pad_kernel<<<cdiv(n, 256), 256, 0, stream>>>(
            W_hh[l], whh_bf[l], N4[l], dh[l], dhP[l], 0x5EED0u + l, 0.5f);  // P_W = 0.5
        awd_bias_kernel<<<cdiv(LDG, 256), 256, 0, stream>>>(b_ih[l], b_hh[l], bsum[l], N4[l], LDG);
    }
    awd_cast_pad_kernel<<<cdiv((long long)NTOK * NINP_P, 256), 256, 0, stream>>>(
        embW, emb_bf, NTOK, NINP, NINP_P, 0u, 1.0f);  // raw tied weights for decoder

    // ---- embedding + input dropout -> x (bf16, K padded) ----
    awd_embed_kernel<<<cdiv((long long)M * NINP_P, 256), 256, 0, stream>>>(tokens, embW, xA);

    bf16_t* xin = xA; bf16_t* xout = xB;
    for (int l = 0; l < 3; ++l) {
        // batched input projection: xg = x @ W_ih^T + (b_ih + b_hh)
        dim3 g1(cdiv(N4[l], 64), cdiv(M, 128));
        awd_gemm_bf16_wmma<4, 2, 2, 2><<<g1, 256, 0, stream>>>(
            xin, dinP[l], wih_bf[l], dinP[l], bsum[l], xg, LDG, M, N4[l], dinP[l]);

        awd_init_state_kernel<<<cdiv((long long)BB * dhP[l], 256), 256, 0, stream>>>(
            h0[l], c0[l], hbf, cst, dh[l], dhP[l]);

        // sequential recurrence: 70 x (skinny WMMA GEMM + gate update)
        dim3 g2(cdiv(N4[l], 256), cdiv(BB, 16));
        for (int t = 0; t < TT; ++t) {
            awd_gemm_bf16_wmma<1, 8, 1, 2><<<g2, 256, 0, stream>>>(
                hbf, dhP[l], whh_bf[l], dhP[l], nullptr, gbf, LDG, BB, N4[l], dhP[l]);
            awd_lstm_gate_kernel<<<cdiv((long long)BB * dhP[l], 256), 256, 0, stream>>>(
                xg + (size_t)t * BB * LDG, gbf, cst, hbf,
                xout + (size_t)t * BB * dhP[l],
                dh[l], dhP[l], LDG, pkeepOut[l], 0x7A57u + l);
        }
        bf16_t* tmp = xin; xin = xout; xout = tmp;
    }

    // ---- tied decoder: out = x @ emb_W^T + dec_b  (5600 x 33278 x 400) ----
    dim3 g3(cdiv(NTOK, 64), cdiv(M, 128));
    awd_gemm_bf16_wmma<4, 2, 2, 2><<<g3, 256, 0, stream>>>(
        xin, NINP_P, emb_bf, NINP_P, dec_b, outp, NTOK, M, NTOK, NINP_P);
}
